// BalancedMoE_6674379178474
// MI455X (gfx1250) — compile-verified, matrix operand order checked by reading, not run on hardware
//
#include <hip/hip_runtime.h>
#include <hip/hip_bf16.h>
#include <math.h>

// ---------------- problem constants ----------------
#define BB   4
#define TT   2048
#define CC   896
#define EE   3
#define HH   3584
#define NN   (BB*TT)              // 8192 tokens
#define CAP  3413                 // int(1.25 * N / E)
#define CAPP 3456                 // CAP rounded up to 64 (54 tiles of 64)
#define MT64 (CAPP/64)            // 54 row tiles per expert
#define NEG_SENT (-1.0e30f)

// padded LDS row strides (ushorts): 128 data + 8 pad = 272B -> bank-conflict free
#define SA_LD 136
#define SB_LD 136

typedef __attribute__((ext_vector_type(16))) __bf16 v16bf;
typedef __attribute__((ext_vector_type(8)))  float  v8f;

union Frag16 { v16bf v; uint4 q[2]; };

__device__ __forceinline__ unsigned short f2bf(float f) {
    union { float f; unsigned u; } v; v.f = f;
    unsigned r = v.u + 0x7FFFu + ((v.u >> 16) & 1u);   // round to nearest even
    return (unsigned short)(r >> 16);
}

__device__ __forceinline__ float gelu_exact(float x) {
    return 0.5f * x * (1.0f + erff(x * 0.70710678118654752f));
}

// A/B fragment from padded LDS (documented CDNA5 16-bit WMMA lane layout):
// lanes 0-15 -> row=lane,     K = {k..k+7,  k+16..k+23}
// lanes 16-31 -> row=lane-16, K = {k+8..k+15, k+24..k+31}
__device__ __forceinline__ v16bf frag_lds(const unsigned short* s, int row, int kk, int kofs, int ld) {
    Frag16 f;
    f.q[0] = *(const uint4*)(s + row * ld + kk + kofs);
    f.q[1] = *(const uint4*)(s + row * ld + kk + kofs + 16);
    return f.v;
}

// ---------------- kernel 0: init workspace ----------------
__global__ void moe_init(float* scal, int* idxl, int nidx) {
    int i = blockIdx.x * blockDim.x + threadIdx.x;
    if (i < 8) scal[i] = 0.0f;                 // zsum, psum[3], dcount[3]
    if (i < nidx) idxl[i] = -1;
}

// ---------------- kernel 1: router (one wave per token) ----------------
__global__ void moe_router(const float* __restrict__ x, const float* __restrict__ Wr,
                           unsigned short* __restrict__ xb, float* __restrict__ scores,
                           float* __restrict__ scal) {
    int wave = threadIdx.x >> 5, lane = threadIdx.x & 31;
    int n = blockIdx.x * 8 + wave;
    if (n >= NN) return;
    const float* xr = x + (size_t)n * CC;
    float s0 = 0.f, s1 = 0.f, s2 = 0.f;
    for (int c = lane; c < CC; c += 32) {
        float xv = xr[c];
        s0 += xv * Wr[c];
        s1 += xv * Wr[CC + c];
        s2 += xv * Wr[2 * CC + c];
        xb[(size_t)n * CC + c] = f2bf(xv);
    }
    for (int off = 16; off; off >>= 1) {
        s0 += __shfl_xor(s0, off, 32);
        s1 += __shfl_xor(s1, off, 32);
        s2 += __shfl_xor(s2, off, 32);
    }
    if (lane == 0) {
        float m  = fmaxf(s0, fmaxf(s1, s2));
        float e0 = expf(s0 - m), e1 = expf(s1 - m), e2 = expf(s2 - m);
        float sum = e0 + e1 + e2;
        float lse = m + logf(sum);
        atomicAdd(&scal[0], lse * lse);
        float p0 = e0 / sum, p1 = e1 / sum, p2 = e2 / sum;
        atomicAdd(&scal[1], p0); atomicAdd(&scal[2], p1); atomicAdd(&scal[3], p2);
        int   am = 0; float pm = p0;
        if (p1 > pm) { am = 1; pm = p1; }
        if (p2 > pm) { am = 2; pm = p2; }
        scores[0 * NN + n] = (am == 0) ? p0 : NEG_SENT;
        scores[1 * NN + n] = (am == 1) ? p1 : NEG_SENT;
        scores[2 * NN + n] = (am == 2) ? p2 : NEG_SENT;
    }
}

// ---------------- kernel 2: weights fp32 -> bf16 ----------------
__global__ void moe_castw(const float* __restrict__ W1, const float* __restrict__ W2,
                          unsigned short* __restrict__ W1b, unsigned short* __restrict__ W2b) {
    size_t i = (size_t)blockIdx.x * blockDim.x + threadIdx.x;
    size_t total = (size_t)EE * HH * CC;
    if (i < total) { W1b[i] = f2bf(W1[i]); W2b[i] = f2bf(W2[i]); }
}

// ---------------- kernel 3: capacity selection via exact rank ----------------
__global__ __launch_bounds__(1024)
void moe_rank(const float* __restrict__ scores, float* __restrict__ gate,
              int* __restrict__ idxl, float* __restrict__ scal) {
    __shared__ float sc[NN];                   // 32 KB
    int e = blockIdx.y;
    const float* se = scores + (size_t)e * NN;
    for (int i = threadIdx.x; i < NN; i += 1024) sc[i] = se[i];
    __syncthreads();
    int n = blockIdx.x * 1024 + threadIdx.x;
    float my = sc[n];
    if (my > -1.0e29f) {                       // token assigned to this expert
        int rank = 0;
        for (int m = 0; m < NN; m += 4) {      // vectorized LDS scan
            float4 v = *(const float4*)(sc + m);
            rank += (v.x > my) || (v.x == my && (m + 0) < n);
            rank += (v.y > my) || (v.y == my && (m + 1) < n);
            rank += (v.z > my) || (v.z == my && (m + 2) < n);
            rank += (v.w > my) || (v.w == my && (m + 3) < n);
        }
        if (rank < CAP) {
            gate[n] = my / (my + 1e-6f);       // top-1: row-sum of gates == p
            idxl[e * CAPP + rank] = n;         // unique slot -> free compaction
            atomicAdd(&scal[4 + e], 1.0f);
        } else {
            gate[n] = 0.0f;                    // dropped by capacity
        }
    }
}

// ---------------- kernel 4: out = x (residual base) ----------------
__global__ void moe_copy(const float4* __restrict__ x4, float4* __restrict__ o4, int n4) {
    int i = blockIdx.x * blockDim.x + threadIdx.x;
    if (i < n4) o4[i] = x4[i];
}

// ---------------- kernel 5: GEMM1  h = gelu(gather(x) @ W1^T), bf16 out ------
// block = 256 (8 waves), tile = 64 rows x 128 cols; wave w: cols [w*16,w*16+16),
// 4 stacked 16x16 WMMA accumulators in M. A and B both staged in padded LDS.
// Single-buffered: compiler overlaps ds fragment loads with WMMAs via partial
// s_wait_dscnt; next-chunk global latency is covered by prefetch + multi-block
// occupancy (52 KB LDS/block on a 320 KB WGP).
__global__ __launch_bounds__(256)
void moe_gemm1(const unsigned short* __restrict__ xb, const unsigned short* __restrict__ W1b,
               const int* __restrict__ idxl, unsigned short* __restrict__ hbuf) {
    __shared__ unsigned short sA[64 * SA_LD];   // ~17 KB
    __shared__ unsigned short sB[128 * SB_LD];  // ~34 KB
    const int e     = blockIdx.z;
    const int mbase = blockIdx.y * 64;
    const int colb0 = blockIdx.x * 128;
    const int wave  = threadIdx.x >> 5, lane = threadIdx.x & 31;
    const int nloc  = lane & 15;
    const int kofs  = (lane >> 4) * 8;
    const unsigned short* Bbase = W1b + (size_t)e * HH * CC;

    // A-stager: 64 rows x 16 segs = 1024 uint4 units, 4 per thread (tokens fixed)
    int   arow[4]; const unsigned short* asrc[4];
    #pragma unroll
    for (int i = 0; i < 4; ++i) {
        int u = i * 256 + threadIdx.x;
        arow[i] = u >> 4;
        int seg = u & 15;
        int tok = idxl[e * CAPP + mbase + arow[i]];
        asrc[i] = (tok >= 0) ? (xb + (size_t)tok * CC + seg * 8) : nullptr;
    }

    v8f acc0 = {}, acc1 = {}, acc2 = {}, acc3 = {};
    for (int kc = 0; kc < CC; kc += 128) {
        #pragma unroll
        for (int i = 0; i < 4; ++i) {          // stage A (gathered, zero-fill)
            int u = i * 256 + threadIdx.x;
            uint4 av = make_uint4(0u, 0u, 0u, 0u);
            if (asrc[i]) av = *(const uint4*)(asrc[i] + kc);
            *(uint4*)(sA + arow[i] * SA_LD + (u & 15) * 8) = av;
        }
        #pragma unroll
        for (int i = 0; i < 8; ++i) {          // stage B (coalesced rows)
            int u = i * 256 + threadIdx.x;
            int col = u >> 4, seg = u & 15;
            const unsigned short* src = Bbase + (size_t)(colb0 + col) * CC + kc + seg * 8;
            if (i == 0 && kc + 128 < CC) __builtin_prefetch(src + 128, 0, 1);
            *(uint4*)(sB + col * SB_LD + seg * 8) = *(const uint4*)src;
        }
        __syncthreads();
        #pragma unroll
        for (int kk = 0; kk < 128; kk += 32) {
            v16bf b = frag_lds(sB, wave * 16 + nloc, kk, kofs, SB_LD);
            v16bf a0 = frag_lds(sA,  0 + nloc, kk, kofs, SA_LD);
            acc0 = __builtin_amdgcn_wmma_f32_16x16x32_bf16(false, a0, false, b, (short)0, acc0, false, false);
            v16bf a1 = frag_lds(sA, 16 + nloc, kk, kofs, SA_LD);
            acc1 = __builtin_amdgcn_wmma_f32_16x16x32_bf16(false, a1, false, b, (short)0, acc1, false, false);
            v16bf a2 = frag_lds(sA, 32 + nloc, kk, kofs, SA_LD);
            acc2 = __builtin_amdgcn_wmma_f32_16x16x32_bf16(false, a2, false, b, (short)0, acc2, false, false);
            v16bf a3 = frag_lds(sA, 48 + nloc, kk, kofs, SA_LD);
            acc3 = __builtin_amdgcn_wmma_f32_16x16x32_bf16(false, a3, false, b, (short)0, acc3, false, false);
        }
        __syncthreads();
    }
    // D layout: lane -> N=lane&15 ; vgpr r -> M = r + 8*(lane>=16)
    const int mhalf8 = (lane >> 4) * 8;
    const int col    = colb0 + wave * 16 + nloc;
    v8f accs[4] = { acc0, acc1, acc2, acc3 };
    #pragma unroll
    for (int ms = 0; ms < 4; ++ms) {
        size_t rbase = (size_t)(e * CAPP + mbase + ms * 16 + mhalf8);
        #pragma unroll
        for (int r = 0; r < 8; ++r)
            hbuf[(rbase + r) * HH + col] = f2bf(gelu_exact(accs[ms][r]));
    }
}

// ---------------- kernel 6: GEMM2  out[token] += gate * (h @ W2^T) ----------
__global__ __launch_bounds__(256)
void moe_gemm2(const unsigned short* __restrict__ hbuf, const unsigned short* __restrict__ W2b,
               const int* __restrict__ idxl, const float* __restrict__ gate,
               float* __restrict__ out) {
    __shared__ unsigned short sA[64 * SA_LD];
    __shared__ unsigned short sB[128 * SB_LD];
    const int e     = blockIdx.z;
    const int mbase = blockIdx.y * 64;
    const int colb0 = blockIdx.x * 128;
    const int wave  = threadIdx.x >> 5, lane = threadIdx.x & 31;
    const int nloc  = lane & 15;
    const int kofs  = (lane >> 4) * 8;
    const unsigned short* Bbase = W2b + (size_t)e * CC * HH;

    v8f acc0 = {}, acc1 = {}, acc2 = {}, acc3 = {};
    for (int kc = 0; kc < HH; kc += 128) {
        #pragma unroll
        for (int i = 0; i < 4; ++i) {          // stage A (dense h rows)
            int u = i * 256 + threadIdx.x;
            int row = u >> 4, seg = u & 15;
            const unsigned short* src = hbuf + ((size_t)(e * CAPP + mbase + row)) * HH + kc + seg * 8;
            *(uint4*)(sA + row * SA_LD + seg * 8) = *(const uint4*)src;
        }
        #pragma unroll
        for (int i = 0; i < 8; ++i) {          // stage B (coalesced rows)
            int u = i * 256 + threadIdx.x;
            int col = u >> 4, seg = u & 15;
            const unsigned short* src = Bbase + (size_t)(colb0 + col) * HH + kc + seg * 8;
            if (i == 0 && kc + 128 < HH) __builtin_prefetch(src + 128, 0, 1);
            *(uint4*)(sB + col * SB_LD + seg * 8) = *(const uint4*)src;
        }
        __syncthreads();
        #pragma unroll
        for (int kk = 0; kk < 128; kk += 32) {
            v16bf b = frag_lds(sB, wave * 16 + nloc, kk, kofs, SB_LD);
            v16bf a0 = frag_lds(sA,  0 + nloc, kk, kofs, SA_LD);
            acc0 = __builtin_amdgcn_wmma_f32_16x16x32_bf16(false, a0, false, b, (short)0, acc0, false, false);
            v16bf a1 = frag_lds(sA, 16 + nloc, kk, kofs, SA_LD);
            acc1 = __builtin_amdgcn_wmma_f32_16x16x32_bf16(false, a1, false, b, (short)0, acc1, false, false);
            v16bf a2 = frag_lds(sA, 32 + nloc, kk, kofs, SA_LD);
            acc2 = __builtin_amdgcn_wmma_f32_16x16x32_bf16(false, a2, false, b, (short)0, acc2, false, false);
            v16bf a3 = frag_lds(sA, 48 + nloc, kk, kofs, SA_LD);
            acc3 = __builtin_amdgcn_wmma_f32_16x16x32_bf16(false, a3, false, b, (short)0, acc3, false, false);
        }
        __syncthreads();
    }
    const int mhalf8 = (lane >> 4) * 8;
    const int col    = colb0 + wave * 16 + nloc;
    v8f accs[4] = { acc0, acc1, acc2, acc3 };
    #pragma unroll
    for (int ms = 0; ms < 4; ++ms) {
        #pragma unroll
        for (int r = 0; r < 8; ++r) {
            int m = mbase + ms * 16 + mhalf8 + r;
            int tok = idxl[e * CAPP + m];
            if (tok >= 0) {
                size_t o = (size_t)tok * CC + col;
                out[o] += gate[tok] * accs[ms][r];   // disjoint (token,col) writes
            }
        }
    }
}

// ---------------- kernel 7: scalar loss ----------------
__global__ void moe_loss(const float* __restrict__ scal, float* __restrict__ out_scalar) {
    if (threadIdx.x == 0 && blockIdx.x == 0) {
        float invN = 1.0f / (float)NN;
        float z = 0.001f * scal[0] * invN;
        float aux = 0.0f;
        for (int e = 0; e < EE; ++e) aux += (scal[4 + e] * invN) * (scal[1 + e] * invN);
        out_scalar[0] = z + 0.1f * (float)EE * aux;
    }
}

// ---------------- host launcher ----------------
static inline size_t alignup(size_t x) { return (x + 255) & ~(size_t)255; }

extern "C" void kernel_launch(void* const* d_in, const int* in_sizes, int n_in,
                              void* d_out, int out_size, void* d_ws, size_t ws_size,
                              hipStream_t stream) {
    const float* x  = (const float*)d_in[0];   // [B,T,C]
    const float* Wr = (const float*)d_in[1];   // [E,C]
    const float* W1 = (const float*)d_in[2];   // [E,H,C]
    const float* W2 = (const float*)d_in[3];   // [E,C,H]
    float* out = (float*)d_out;                // [B,T,C] floats + 1 scalar

    char* ws = (char*)d_ws;
    size_t off = 0;
    float* scal = (float*)(ws + off);           off = alignup(off + 8 * sizeof(float));
    float* scores = (float*)(ws + off);         off = alignup(off + (size_t)EE * NN * sizeof(float));
    float* gate = (float*)(ws + off);           off = alignup(off + (size_t)NN * sizeof(float));
    int*   idxl = (int*)(ws + off);             off = alignup(off + (size_t)EE * CAPP * sizeof(int));
    unsigned short* xb  = (unsigned short*)(ws + off); off = alignup(off + (size_t)NN * CC * 2);
    unsigned short* W1b = (unsigned short*)(ws + off); off = alignup(off + (size_t)EE * HH * CC * 2);
    unsigned short* W2b = (unsigned short*)(ws + off); off = alignup(off + (size_t)EE * HH * CC * 2);
    unsigned short* hbuf = (unsigned short*)(ws + off); off = alignup(off + (size_t)EE * CAPP * HH * 2);
    (void)ws_size; (void)in_sizes; (void)n_in; (void)out_size;

    // 0) init scratch state (ws is NOT re-poisoned between replays)
    {
        int nidx = EE * CAPP;
        moe_init<<<(nidx + 255) / 256, 256, 0, stream>>>(scal, idxl, nidx);
    }
    // 1) router + bf16 cast of x
    moe_router<<<NN / 8, 256, 0, stream>>>(x, Wr, xb, scores, scal);
    // 2) bf16 cast of weights
    {
        size_t total = (size_t)EE * HH * CC;
        moe_castw<<<(unsigned)((total + 255) / 256), 256, 0, stream>>>(W1, W2, W1b, W2b);
    }
    // 3) capacity selection (exact per-expert rank)
    {
        dim3 g(NN / 1024, EE);
        moe_rank<<<g, 1024, 0, stream>>>(scores, gate, idxl, scal);
    }
    // 4) residual base: out = x
    {
        int n4 = NN * CC / 4;
        moe_copy<<<(n4 + 255) / 256, 256, 0, stream>>>((const float4*)x, (float4*)out, n4);
    }
    // 5) expert up-projection + GELU (bf16 WMMA, 64x128 tiles)
    {
        dim3 g(HH / 128, MT64, EE);
        moe_gemm1<<<g, 256, 0, stream>>>(xb, W1b, idxl, hbuf);
    }
    // 6) expert down-projection + gated combine (bf16 WMMA, 64x128 tiles)
    {
        dim3 g(CC / 128, MT64, EE);
        moe_gemm2<<<g, 256, 0, stream>>>(hbuf, W2b, idxl, gate, out);
    }
    // 7) aux + z loss scalar
    moe_loss<<<1, 64, 0, stream>>>(scal, out + (size_t)NN * CC);
}
